// GAT_27960237097141
// MI455X (gfx1250) — matrix operand order repaired and checked by reading, not compiled, not run on hardware
//
#include <hip/hip_runtime.h>
#include <hip/hip_bf16.h>

// ---------------------------------------------------------------------------
// GAT 2-layer forward for MI455X (gfx1250, wave32, WMMA).
// Pipeline: pack operands to bf16 in WMMA lane layout (streaming pass), then
// strip-mined v_wmma_f32_16x16x32_bf16 GEMM (contiguous b128 operand loads,
// A read exactly once), then bandwidth-bound edge softmax/aggregation with
// coalesced global float atomics.
// ---------------------------------------------------------------------------

#define N_NODES 100000
#define N_EDGES 1600000
#define IN_DIM  128      // K for both GEMMs
#define KSTEPS  4        // 128 / 32
#define F1      128      // heads*hid = 4*32
#define H1      4
#define D1      32
#define F2      64       // layer2 out dim (1 head x 64)
#define NEG_SLOPE 0.2f

typedef __attribute__((ext_vector_type(16))) __bf16          v16bf;
typedef __attribute__((ext_vector_type(8)))  float           v8f;
typedef __attribute__((ext_vector_type(16))) unsigned short  v16us;

// ---- monotone float <-> uint encoding for atomic max on signed floats ----
__device__ __forceinline__ unsigned fenc(float f) {
    unsigned u = __float_as_uint(f);
    return (u & 0x80000000u) ? ~u : (u | 0x80000000u);
}
__device__ __forceinline__ float fdec(unsigned u) {
    u = (u & 0x80000000u) ? (u & 0x7fffffffu) : ~u;
    return __uint_as_float(u);
}
__device__ __forceinline__ unsigned short f2bf(float f) {
    unsigned u = __float_as_uint(f);
    return (unsigned short)((u + 0x7fffu + ((u >> 16) & 1u)) >> 16); // RNE
}

// ---------------------------------------------------------------------------
// Operand packing.  A-operand lane layout (16-bit A 16x32, contiguous k):
//   lanes 0-15  hold row M=lane,    K = kk+0..15
//   lanes 16-31 hold row M=lane-16, K = kk+16..31
// B-operand (32x16): lanes 0-15 hold col N=lane,    K = kk+0..15
//                    lanes 16-31 hold col N=lane-16, K = kk+16..31
// Packed element index = ((tile*KSTEPS + j)*32 + lane)*16 + i  (bf16)
// ---------------------------------------------------------------------------
__global__ void pack_a(const float* __restrict__ A, unsigned short* __restrict__ PA,
                       int nRows) {
    int t = blockIdx.x * blockDim.x + threadIdx.x;        // (rowTile, j, lane)
    int total = (nRows / 16) * KSTEPS * 32;
    if (t >= total) return;
    int lane    = t & 31;
    int j       = (t >> 5) & (KSTEPS - 1);
    int rowTile = t >> 7;
    int laneLo = lane & 15, laneHi = lane >> 4;
    const float* arow = A + (size_t)(rowTile * 16 + laneLo) * IN_DIM + j * 32;
    unsigned short* o = PA + (size_t)t * 16;
#pragma unroll
    for (int i = 0; i < 8; ++i) {
        o[i]     = f2bf(arow[laneHi * 16 + i]);
        o[i + 8] = f2bf(arow[laneHi * 16 + 8 + i]);
    }
}

template <int FCOLS>
__global__ void pack_b(const float* __restrict__ B, unsigned short* __restrict__ PB) {
    int t = blockIdx.x * blockDim.x + threadIdx.x;        // (colTile, j, lane)
    int total = (FCOLS / 16) * KSTEPS * 32;
    if (t >= total) return;
    int lane    = t & 31;
    int j       = (t >> 5) & (KSTEPS - 1);
    int colTile = t >> 7;
    int laneLo = lane & 15, laneHi = lane >> 4;
    unsigned short* o = PB + (size_t)t * 16;
#pragma unroll
    for (int i = 0; i < 16; ++i) {
        o[i] = f2bf(B[(size_t)(j * 32 + laneHi * 16 + i) * FCOLS + colTile * 16 + laneLo]);
    }
}

// ---------------------------------------------------------------------------
// Strip-mined GEMM: one wave -> 16-row x FCOLS strip of C.
// Per K-step: 1 packed-A fragment load feeds FCOLS/16 WMMAs.
// ---------------------------------------------------------------------------
template <int FCOLS>
__global__ void gat_gemm_wmma(const unsigned short* __restrict__ PA,
                              const unsigned short* __restrict__ PB,
                              float* __restrict__ C, int nRows) {
    constexpr int COLT = FCOLS / 16;
    const int lane = threadIdx.x & 31;
    const int wave = threadIdx.x >> 5;
    const int rowTile = blockIdx.x * (blockDim.x >> 5) + wave;
    if (rowTile >= nRows / 16) return;      // wave-uniform; EXEC all-ones inside

    const unsigned short* pa = PA + ((size_t)rowTile * KSTEPS * 32 + lane) * 16;
    const unsigned short* pb = PB + (size_t)lane * 16;

    v8f acc[COLT];
#pragma unroll
    for (int n = 0; n < COLT; ++n) acc[n] = (v8f){};

#pragma unroll
    for (int j = 0; j < KSTEPS; ++j) {
        v16bf av = __builtin_bit_cast(
            v16bf, *(const v16us*)(pa + (size_t)j * 32 * 16));
#pragma unroll
        for (int n = 0; n < COLT; ++n) {
            v16bf bv = __builtin_bit_cast(
                v16bf, *(const v16us*)(pb + (size_t)(n * KSTEPS + j) * 32 * 16));
            acc[n] = __builtin_amdgcn_wmma_f32_16x16x32_bf16(
                false, av, false, bv, (short)0, acc[n], false, false);
        }
    }
    // D layout: vgpr r, lanes 0-15 -> C[r, lane]; lanes 16-31 -> C[8+r, lane-16]
    const int laneLo = lane & 15, laneHi = lane >> 4;
#pragma unroll
    for (int n = 0; n < COLT; ++n) {
#pragma unroll
        for (int r = 0; r < 8; ++r) {
            C[(size_t)(rowTile * 16 + laneHi * 8 + r) * FCOLS + n * 16 + laneLo] =
                acc[n][r];
        }
    }
}

// ---------------------------------------------------------------------------
// Per-node attention logits: el[n,h] = sum_d h[n,h,d]*al[h,d]; er likewise.
// ---------------------------------------------------------------------------
__global__ void node_logits(const float* __restrict__ hfeat,
                            const float* __restrict__ al,
                            const float* __restrict__ ar,
                            float* __restrict__ el, float* __restrict__ er,
                            int n, int H, int D) {
    int t = blockIdx.x * blockDim.x + threadIdx.x;
    if (t >= n * H) return;
    int node = t / H, hh = t % H;
    const float* hp = hfeat + (size_t)node * H * D + (size_t)hh * D;
    const float* ap = al + hh * D;
    const float* bp = ar + hh * D;
    float sl = 0.f, sr = 0.f;
    for (int d = 0; d < D; ++d) {
        float v = hp[d];
        sl = fmaf(v, ap[d], sl);
        sr = fmaf(v, bp[d], sr);
    }
    el[t] = sl;
    er[t] = sr;
}

// Pass 1: e = leaky_relu(el[src]+er[dst]); atomic max per (dst,h).
__global__ void edge_logits_max(const int* __restrict__ src,
                                const int* __restrict__ dst,
                                const float* __restrict__ el,
                                const float* __restrict__ er,
                                float* __restrict__ eval,
                                unsigned* __restrict__ menc, int E, int H) {
    int e = blockIdx.x * blockDim.x + threadIdx.x;
    if (e >= E) return;
    int s = src[e], d = dst[e];
    for (int h = 0; h < H; ++h) {
        float v = el[s * H + h] + er[d * H + h];
        v = v > 0.f ? v : NEG_SLOPE * v;
        eval[(size_t)e * H + h] = v;
        atomicMax(&menc[d * H + h], fenc(v));
    }
}

// Pass 2: ex = exp(e - m[dst]); atomic sum per (dst,h).  (in-place on eval)
__global__ void edge_exp_sum(const int* __restrict__ dst,
                             float* __restrict__ eval,
                             const unsigned* __restrict__ menc,
                             float* __restrict__ ssum, int E, int H) {
    int e = blockIdx.x * blockDim.x + threadIdx.x;
    if (e >= E) return;
    int d = dst[e];
    for (int h = 0; h < H; ++h) {
        float m  = fdec(menc[d * H + h]);
        float x  = __expf(eval[(size_t)e * H + h] - m);
        eval[(size_t)e * H + h] = x;
        atomicAdd(&ssum[d * H + h], x);
    }
}

// Pass 3: out[dst,h,:] += h[src,h,:] * (ex/s[dst,h]).  One thread per (edge,d).
__global__ void edge_scatter(const int* __restrict__ src,
                             const int* __restrict__ dst,
                             const float* __restrict__ hfeat,
                             const float* __restrict__ ex,
                             const float* __restrict__ ssum,
                             float* __restrict__ out, int E, int H, int D) {
    long long t = (long long)blockIdx.x * blockDim.x + threadIdx.x;
    int e  = (int)(t / D);
    int dd = (int)(t % D);
    if (e >= E) return;
    int s = src[e], d = dst[e];
    for (int h = 0; h < H; ++h) {
        float alpha = ex[(size_t)e * H + h] / ssum[d * H + h];
        float v = hfeat[((size_t)s * H + h) * D + dd] * alpha;
        atomicAdd(&out[((size_t)d * H + h) * D + dd], v);
    }
}

// SELU elementwise, in place.
__global__ void selu_inplace(float* __restrict__ x, int n) {
    int t = blockIdx.x * blockDim.x + threadIdx.x;
    if (t >= n) return;
    const float scale = 1.0507009873554805f;
    const float alpha = 1.6732632423543772f;
    float v = x[t];
    x[t] = scale * (v > 0.f ? v : alpha * (__expf(v) - 1.f));
}

// ---------------------------------------------------------------------------
extern "C" void kernel_launch(void* const* d_in, const int* in_sizes, int n_in,
                              void* d_out, int out_size, void* d_ws, size_t ws_size,
                              hipStream_t stream) {
    (void)in_sizes; (void)n_in; (void)out_size; (void)ws_size;
    const float* x   = (const float*)d_in[0];
    const int*   src = (const int*)  d_in[1];
    const int*   dst = (const int*)  d_in[2];
    const float* W1  = (const float*)d_in[3];
    const float* al1 = (const float*)d_in[4];
    const float* ar1 = (const float*)d_in[5];
    const float* W2  = (const float*)d_in[6];
    const float* al2 = (const float*)d_in[7];
    const float* ar2 = (const float*)d_in[8];
    float* out = (float*)d_out;

    // ---- workspace layout ----
    char* ws = (char*)d_ws;
    unsigned short* PA  = (unsigned short*)ws;                     // N*128 bf16 (reused L2)
    ws += (size_t)N_NODES * IN_DIM * 2;
    unsigned short* PB1 = (unsigned short*)ws; ws += (size_t)IN_DIM * F1 * 2;
    unsigned short* PB2 = (unsigned short*)ws; ws += (size_t)IN_DIM * F2 * 2;
    float*    h1   = (float*)ws;  ws += (size_t)N_NODES * F1 * 4;  // layer1 features
    float*    acc1 = (float*)ws;  ws += (size_t)N_NODES * F1 * 4;  // layer1 agg -> h2
    float*    ex1  = (float*)ws;  ws += (size_t)N_EDGES * H1 * 4;  // logits->exp (reused L2)
    float*    el1  = (float*)ws;  ws += (size_t)N_NODES * H1 * 4;
    float*    er1  = (float*)ws;  ws += (size_t)N_NODES * H1 * 4;
    unsigned* m1   = (unsigned*)ws; ws += (size_t)N_NODES * H1 * 4;
    float*    s1   = (float*)ws;  ws += (size_t)N_NODES * H1 * 4;
    float*    g2   = h1;  // layer2 features (N*64) reuse h1 region

    const int rowTiles = N_NODES / 16;                     // 6250 (exact)

    // ================= Layer 1 =================
    hipMemsetAsync(m1,   0, (size_t)N_NODES * H1 * 4, stream); // enc 0 < enc(-inf)
    hipMemsetAsync(s1,   0, (size_t)N_NODES * H1 * 4, stream);
    hipMemsetAsync(acc1, 0, (size_t)N_NODES * F1 * 4, stream);

    pack_b<F1><<<((F1 / 16) * KSTEPS * 32 + 255) / 256, 256, 0, stream>>>(W1, PB1);
    pack_b<F2><<<((F2 / 16) * KSTEPS * 32 + 255) / 256, 256, 0, stream>>>(W2, PB2);
    {
        int t = rowTiles * KSTEPS * 32;                    // 800k
        pack_a<<<(t + 255) / 256, 256, 0, stream>>>(x, PA, N_NODES);
    }
    gat_gemm_wmma<F1><<<(rowTiles + 3) / 4, 128, 0, stream>>>(PA, PB1, h1, N_NODES);
    {
        int t = N_NODES * H1;
        node_logits<<<(t + 255) / 256, 256, 0, stream>>>(h1, al1, ar1, el1, er1,
                                                         N_NODES, H1, D1);
    }
    edge_logits_max<<<(N_EDGES + 255) / 256, 256, 0, stream>>>(src, dst, el1, er1,
                                                               ex1, m1, N_EDGES, H1);
    edge_exp_sum<<<(N_EDGES + 255) / 256, 256, 0, stream>>>(dst, ex1, m1, s1,
                                                            N_EDGES, H1);
    {
        long long t = (long long)N_EDGES * D1;             // 51.2M
        edge_scatter<<<(unsigned)((t + 255) / 256), 256, 0, stream>>>(
            src, dst, h1, ex1, s1, acc1, N_EDGES, H1, D1);
    }
    {   // h2 = selu(acc1), in place
        int t = N_NODES * F1;
        selu_inplace<<<(t + 255) / 256, 256, 0, stream>>>(acc1, t);
    }

    // ================= Layer 2 (H=1, D=64) =================
    hipMemsetAsync(m1,  0, (size_t)N_NODES * 4, stream);
    hipMemsetAsync(s1,  0, (size_t)N_NODES * 4, stream);
    hipMemsetAsync(out, 0, (size_t)N_NODES * F2 * 4, stream);

    {
        int t = rowTiles * KSTEPS * 32;
        pack_a<<<(t + 255) / 256, 256, 0, stream>>>(acc1, PA, N_NODES);
    }
    gat_gemm_wmma<F2><<<(rowTiles + 3) / 4, 128, 0, stream>>>(PA, PB2, g2, N_NODES);
    node_logits<<<(N_NODES + 255) / 256, 256, 0, stream>>>(g2, al2, ar2, el1, er1,
                                                           N_NODES, 1, F2);
    edge_logits_max<<<(N_EDGES + 255) / 256, 256, 0, stream>>>(src, dst, el1, er1,
                                                               ex1, m1, N_EDGES, 1);
    edge_exp_sum<<<(N_EDGES + 255) / 256, 256, 0, stream>>>(dst, ex1, m1, s1,
                                                            N_EDGES, 1);
    {
        long long t = (long long)N_EDGES * F2;             // 102.4M
        edge_scatter<<<(unsigned)((t + 255) / 256), 256, 0, stream>>>(
            src, dst, g2, ex1, s1, out, N_EDGES, 1, F2);
    }
}
